// LSTMForecaster_31825707663510
// MI455X (gfx1250) — compile-verified, matrix operand order checked by reading, not run on hardware
//
#include <hip/hip_runtime.h>

#define B_ 256
#define T_ 256
#define D_ 64
#define H_ 512
#define G4H_ 2048
#define WPAD 520          // 512 + 8 bf16 row pad -> conflict-minimal ds_load_b128
#define NGRP 8            // batch groups of 32 rows
#define NCS  8            // column-slice WGs per group (64 h-cols each)
#define WAVES_PER_GROUP 64  // 8 WGs * 8 waves

typedef __attribute__((ext_vector_type(16))) __bf16 v16bf;
typedef __attribute__((ext_vector_type(8)))  float  v8f;

union Frag { v16bf v; uint4 q[2]; };

__device__ __forceinline__ unsigned short f2bf(float f) {
  union { float f; unsigned int u; } cv; cv.f = f;
  unsigned int u = cv.u;
  u += 0x7fffu + ((u >> 16) & 1u);   // round-to-nearest-even
  return (unsigned short)(u >> 16);
}
__device__ __forceinline__ float sigmoidf_(float x) {
  return 1.0f / (1.0f + __expf(-x));
}
__device__ __forceinline__ float tanh_(float x) {
  float e = __expf(2.0f * x);
  return (e - 1.0f) / (e + 1.0f);
}

__global__ void cvt_bf16_kernel(const float* __restrict__ src,
                                unsigned short* __restrict__ dst, int n) {
  int i = blockIdx.x * blockDim.x + threadIdx.x;
  if (i < n) dst[i] = f2bf(src[i]);
}

__global__ void zero_u32_kernel(unsigned int* __restrict__ p, int n) {
  int i = blockIdx.x * blockDim.x + threadIdx.x;
  if (i < n) p[i] = 0u;
}

// grid = NGRP*NCS workgroups of 256 threads (8 wave32).
//   blockIdx = grp*NCS + cs ; WG owns rows [grp*32, grp*32+32) and
//   h-columns [cs*64, cs*64+64)  (gate columns g*512 + cs*64 .. +64).
// Whh slice (256 rows x 512 K, bf16, padded) is LDS-resident for all T steps,
// staged once via GLOBAL_LOAD_ASYNC_TO_LDS_B128 (ASYNCcnt, no VGPR round-trip).
// Wave w: mhalf = w&1 (16-row tile), hq = w>>1 (16 h-cols); 4 gate accumulators.
// Cross-WG h exchange: hshare[2][NGRP][32][512] bf16 in L2 + per-group
// monotonic counter (one release-add per wave per step; acquire-spin target 64*t).
__global__ __launch_bounds__(256, 1)
void lstm_layer_kernel(const unsigned short* __restrict__ xin,
                       int t_stride, int b_stride, int kin,
                       const unsigned short* __restrict__ Wih,   // [2048, kin] bf16
                       const unsigned short* __restrict__ Whh,   // [2048, 512] bf16
                       const float* __restrict__ bih,
                       const float* __restrict__ bhh,
                       unsigned short* __restrict__ hshare,      // [2][NGRP][32][512] bf16
                       int* __restrict__ cnt,                    // [NGRP] zeroed
                       unsigned short* __restrict__ hout,        // [T,B,512] bf16 or null
                       float* __restrict__ hlast)                // [B,512] f32 or null
{
  __shared__ __align__(16) unsigned short wlds[256 * WPAD];   // Whh slice, 260KB

  const int tid   = threadIdx.x;
  const int lane  = tid & 31;
  const int wv    = tid >> 5;
  const int grp   = blockIdx.x / NCS;
  const int cs    = blockIdx.x % NCS;
  const int mhalf = wv & 1;           // which 16-row tile of the 32-row group
  const int hq    = wv >> 1;          // which 16-col h-tile of the 64-col slice
  const int lrow  = lane & 15;        // A row / B,D column within tile
  const int khalf = (lane >> 4) * 8;  // per-lane-half K sub-offset (bf16 elems)

  // ---- stage Whh slice into LDS once, via async DMA (global -> LDS direct) ----
  // local row q in [0,256): global row = (q>>6)*512 + cs*64 + (q&63)
  // each lane copies 16B chunks: row q has 64 chunks, lanes cover {lane, lane+32}
  for (int q = wv; q < 256; q += 8) {
    int grow = (q >> 6) * H_ + cs * 64 + (q & 63);
    const unsigned short* grow_p = Whh + (size_t)grow * H_;
    unsigned ldsrow = (unsigned)(size_t)(&wlds[q * WPAD]);
#pragma unroll
    for (int half = 0; half < 2; ++half) {
      int j = lane + half * 32;                 // 16B chunk index within row
      unsigned ldsa = ldsrow + (unsigned)(j * 16);
      const unsigned short* gp = grow_p + j * 8;
      asm volatile("global_load_async_to_lds_b128 %0, %1, off"
                   :: "v"(ldsa), "v"(gp) : "memory");
    }
  }
  asm volatile("s_wait_asynccnt 0" ::: "memory");

  // per-lane gate biases (column = g*512 + cs*64 + hq*16 + lrow)
  float bias[4];
#pragma unroll
  for (int g = 0; g < 4; ++g) {
    int col = g * H_ + cs * 64 + hq * 16 + lrow;
    bias[g] = bih[col] + bhh[col];
  }

  v8f cst = {};             // cell state for this wave's 16x16 h-tile
  int* cntp = cnt + grp;
  const int brow = grp * 32 + mhalf * 16 + lrow;   // A-operand global row
  const int lrowq = hq * 16 + lrow;                // 0..63 within slice

  __syncthreads();          // LDS weights ready (read-only afterwards)

  const int kin_steps = kin >> 5;

#pragma unroll 1
  for (int t = 0; t < T_; ++t) {
    const int cur = t & 1, nxt = cur ^ 1;

    v8f acc[4];
#pragma unroll
    for (int g = 0; g < 4; ++g) {
      v8f z = {};
      acc[g] = z + bias[g];
    }

    // ---- input contribution (no recurrence dep): issued before the spin ----
    {
      const unsigned short* arow =
          xin + (size_t)t * t_stride + (size_t)brow * b_stride + khalf;
#pragma unroll 1
      for (int kt = 0; kt < kin_steps; ++kt) {
        Frag a;
        a.q[0] = *(const uint4*)(arow + kt * 32);
        a.q[1] = *(const uint4*)(arow + kt * 32 + 16);
#pragma unroll
        for (int g = 0; g < 4; ++g) {
          const unsigned short* wrow =
              Wih + (size_t)(g * H_ + cs * 64 + lrowq) * kin + kt * 32 + khalf;
          Frag b;
          b.q[0] = *(const uint4*)(wrow);
          b.q[1] = *(const uint4*)(wrow + 16);
          acc[g] = __builtin_amdgcn_wmma_f32_16x16x32_bf16(
              false, a.v, false, b.v, (short)0, acc[g], false, false);
        }
      }
    }

    // ---- wait until h(t-1) from all WGs of this group is visible ----
    if (t > 0) {
      const int target = WAVES_PER_GROUP * t;
      while (__hip_atomic_load(cntp, __ATOMIC_ACQUIRE,
                               __HIP_MEMORY_SCOPE_AGENT) < target)
        __builtin_amdgcn_s_sleep(2);
    }

    // ---- recurrent contribution: A = h(t-1) from hshare, B = LDS Whh slice ----
    {
      const unsigned short* arow =
          hshare + ((size_t)(cur * NGRP + grp) * 32 + mhalf * 16 + lrow) * H_ + khalf;
#pragma unroll 1
      for (int kt = 0; kt < 16; ++kt) {
        Frag a;
        a.q[0] = *(const uint4*)(arow + kt * 32);
        a.q[1] = *(const uint4*)(arow + kt * 32 + 16);
#pragma unroll
        for (int g = 0; g < 4; ++g) {
          const unsigned short* wrow =
              &wlds[(g * 64 + lrowq) * WPAD + kt * 32 + khalf];
          Frag b;
          b.q[0] = *(const uint4*)(wrow);
          b.q[1] = *(const uint4*)(wrow + 16);
          acc[g] = __builtin_amdgcn_wmma_f32_16x16x32_bf16(
              false, a.v, false, b.v, (short)0, acc[g], false, false);
        }
      }
    }

    // ---- gates -> c,h update; publish h slice; signal ----
#pragma unroll
    for (int r = 0; r < 8; ++r) {
      float iv = sigmoidf_(acc[0][r]);
      float fv = sigmoidf_(acc[1][r]);
      float gv = tanh_(acc[2][r]);
      float ov = sigmoidf_(acc[3][r]);
      float cc = fv * cst[r] + iv * gv;
      cst[r] = cc;
      float hv = ov * tanh_(cc);

      int m   = mhalf * 16 + r + ((lane >> 4) << 3);  // row within 32-row group
      int col = cs * 64 + hq * 16 + lrow;             // h column 0..511
      unsigned short hb = f2bf(hv);
      hshare[((size_t)(nxt * NGRP + grp) * 32 + m) * H_ + col] = hb;
      if (hout)
        hout[((size_t)t * B_ + grp * 32 + m) * H_ + col] = hb;
      if (hlast && t == T_ - 1)
        hlast[(size_t)(grp * 32 + m) * H_ + col] = hv;
    }

    if (lane == 0)
      __hip_atomic_fetch_add(cntp, 1, __ATOMIC_RELEASE, __HIP_MEMORY_SCOPE_AGENT);
  }
}

__global__ void fc_kernel(const float* __restrict__ hlast,
                          const float* __restrict__ fw,
                          const float* __restrict__ fb,
                          float* __restrict__ out) {
  int b = blockIdx.x * blockDim.x + threadIdx.x;
  if (b >= B_) return;
  float s = 0.0f;
  for (int k = 0; k < H_; ++k) s += hlast[b * H_ + k] * fw[k];
  out[b] = s + fb[0];
}

extern "C" void kernel_launch(void* const* d_in, const int* in_sizes, int n_in,
                              void* d_out, int out_size, void* d_ws, size_t ws_size,
                              hipStream_t stream) {
  const float* x    = (const float*)d_in[0];
  const float* Wih0 = (const float*)d_in[1];
  const float* Whh0 = (const float*)d_in[2];
  const float* bih0 = (const float*)d_in[3];
  const float* bhh0 = (const float*)d_in[4];
  const float* Wih1 = (const float*)d_in[5];
  const float* Whh1 = (const float*)d_in[6];
  const float* bih1 = (const float*)d_in[7];
  const float* bhh1 = (const float*)d_in[8];
  const float* fcw  = (const float*)d_in[9];
  const float* fcb  = (const float*)d_in[10];

  char* ws = (char*)d_ws;
  size_t off = 0;
  auto take = [&](size_t bytes) -> char* {
    char* p = ws + off;
    off = (off + bytes + 255) & ~(size_t)255;
    return p;
  };

  unsigned short* xbf   = (unsigned short*)take((size_t)B_ * T_ * D_ * 2);
  unsigned short* wih0b = (unsigned short*)take((size_t)G4H_ * D_ * 2);
  unsigned short* whh0b = (unsigned short*)take((size_t)G4H_ * H_ * 2);
  unsigned short* wih1b = (unsigned short*)take((size_t)G4H_ * H_ * 2);
  unsigned short* whh1b = (unsigned short*)take((size_t)G4H_ * H_ * 2);
  unsigned short* h1bf  = (unsigned short*)take((size_t)T_ * B_ * H_ * 2);
  float*          hlast = (float*)take((size_t)B_ * H_ * 4);

  // sync region (zeroed every launch): [cnt0 8][cnt1 8][pad][hshare0][hshare1]
  const size_t hshare_bytes = (size_t)2 * NGRP * 32 * H_ * 2;   // 512 KB each
  char* syncbase = take(256 + 2 * hshare_bytes);
  int* cnt0 = (int*)syncbase;
  int* cnt1 = (int*)(syncbase + 64);
  unsigned short* hshare0 = (unsigned short*)(syncbase + 256);
  unsigned short* hshare1 = (unsigned short*)(syncbase + 256 + hshare_bytes);

  int n;
  n = B_ * T_ * D_;
  cvt_bf16_kernel<<<(n + 255) / 256, 256, 0, stream>>>(x, xbf, n);
  n = G4H_ * D_;
  cvt_bf16_kernel<<<(n + 255) / 256, 256, 0, stream>>>(Wih0, wih0b, n);
  n = G4H_ * H_;
  cvt_bf16_kernel<<<(n + 255) / 256, 256, 0, stream>>>(Whh0, whh0b, n);
  cvt_bf16_kernel<<<(n + 255) / 256, 256, 0, stream>>>(Wih1, wih1b, n);
  cvt_bf16_kernel<<<(n + 255) / 256, 256, 0, stream>>>(Whh1, whh1b, n);

  n = (int)((256 + 2 * hshare_bytes) / 4);
  zero_u32_kernel<<<(n + 255) / 256, 256, 0, stream>>>((unsigned int*)syncbase, n);

  // Layer 0: x is [B, T, D] -> row(t,b) offset = t*D + b*(T*D)
  lstm_layer_kernel<<<NGRP * NCS, 256, 0, stream>>>(
      xbf, D_, T_ * D_, D_, wih0b, whh0b, bih0, bhh0,
      hshare0, cnt0, h1bf, nullptr);

  // Layer 1: h1 is [T, B, H] -> row(t,b) offset = t*(B*H) + b*H
  lstm_layer_kernel<<<NGRP * NCS, 256, 0, stream>>>(
      h1bf, B_ * H_, H_, H_, wih1b, whh1b, bih1, bhh1,
      hshare1, cnt1, nullptr, hlast);

  fc_kernel<<<1, 256, 0, stream>>>(hlast, fcw, fcb, (float*)d_out);
}
